// EstimatorCRF_39702677684376
// MI455X (gfx1250) — matrix operand order repaired and checked by reading, not compiled, not run on hardware
//
#include <hip/hip_runtime.h>
#include <hip/hip_bf16.h>

#define EDIM 512
#define HDIM 512
#define G3   1536
#define D2   1024
#define SLEN 2048
#define TLEN 2048

typedef __bf16 bf16_t;
typedef __attribute__((ext_vector_type(8)))  __bf16 v8bf;
typedef __attribute__((ext_vector_type(16))) __bf16 v16bf;
typedef __attribute__((ext_vector_type(8)))  float  v8f;

// ---------------- WMMA fragment helpers (CDNA5 wave32 layouts, CDNA5 ISA) ----------------
// 16-bit A matrix 16x32 (MxK): lanes 0-15 -> M=lane, K {0..7, 16..23}; lanes 16-31 -> M=lane-16, K {8..15, 24..31}
__device__ __forceinline__ v16bf load_frag_a(const bf16_t* A, int lda, int lane) {
  int r  = lane & 15;
  int ks = (lane >> 4) << 3;
  const bf16_t* p = A + r * lda + ks;
  v8bf lo = *(const v8bf*)(p);
  v8bf hi = *(const v8bf*)(p + 16);
  return __builtin_shufflevector(lo, hi, 0,1,2,3,4,5,6,7,8,9,10,11,12,13,14,15);
}

// B matrix 32x16 (KxN) sourced from a row-major [N x K] array (i.e. B^T):
// lanes 0-15 -> N=lane, K {0..7, 16..23}; lanes 16-31 -> N=lane-16, K {8..15, 24..31}
__device__ __forceinline__ v16bf load_frag_b(const bf16_t* B, int ldb, int lane) {
  int n  = lane & 15;
  int ks = (lane >> 4) << 3;
  const bf16_t* p = B + n * ldb + ks;
  v8bf lo = *(const v8bf*)(p);
  v8bf hi = *(const v8bf*)(p + 16);
  return __builtin_shufflevector(lo, hi, 0,1,2,3,4,5,6,7,8,9,10,11,12,13,14,15);
}

__device__ __forceinline__ v8f wmma_bf16(v16bf a, v16bf b, v8f c) {
  // (neg_a, A, neg_b, B, c_mod, C, reuse_a, reuse_b)
  return __builtin_amdgcn_wmma_f32_16x16x32_bf16(false, a, false, b, (short)0, c, false, false);
}

// C/D 16x16 f32: vgpr r, lane l -> M = r + 8*(l>>4), N = l&15
__device__ __forceinline__ void store_tile(float* C, int ldc, int m0, int n0,
                                           v8f acc, const float* bias, float alpha, int lane) {
  int n  = n0 + (lane & 15);
  int mb = m0 + ((lane >> 4) << 3);
  float bv = bias ? bias[n] : 0.0f;
#pragma unroll
  for (int r = 0; r < 8; ++r)
    C[(size_t)(mb + r) * ldc + n] = acc[r] * alpha + bv;
}

// ---------------- generic C = alpha * (A[MxK] . B[NxK]^T) + bias ----------------
// one wave per 32x32 macro-tile (2x2 WMMA fragments), bf16 in / f32 out
__global__ __launch_bounds__(256) void wmma_gemm_bf16_kernel(
    const bf16_t* __restrict__ A, const bf16_t* __restrict__ B,
    float* __restrict__ C, const float* __restrict__ bias,
    int M, int N, int K, int lda, int ldb, int ldc, float alpha)
{
  int lane = threadIdx.x & 31;
  int tIdx = blockIdx.x * (blockDim.x >> 5) + (threadIdx.x >> 5);
  int Nt = N >> 5, Mt = M >> 5;
  if (tIdx >= Mt * Nt) return;              // wave-uniform exit: EXEC stays all-ones for WMMA
  int tm = tIdx / Nt, tn = tIdx % Nt;
  int m0 = tm << 5, n0 = tn << 5;

  v8f acc00 = {}, acc01 = {}, acc10 = {}, acc11 = {};
  const bf16_t* Abase0 = A + (size_t)m0 * lda;
  const bf16_t* Abase1 = A + (size_t)(m0 + 16) * lda;
  const bf16_t* Bbase0 = B + (size_t)n0 * ldb;
  const bf16_t* Bbase1 = B + (size_t)(n0 + 16) * ldb;

  for (int k0 = 0; k0 < K; k0 += 32) {
    if (k0 + 32 < K) {                       // gfx1250 global_prefetch of next K-slab
      __builtin_prefetch(Abase0 + (lane & 15) * lda + k0 + 32, 0, 1);
      __builtin_prefetch(Bbase0 + (lane & 15) * ldb + k0 + 32, 0, 1);
    }
    v16bf a0 = load_frag_a(Abase0 + k0, lda, lane);
    v16bf a1 = load_frag_a(Abase1 + k0, lda, lane);
    v16bf b0 = load_frag_b(Bbase0 + k0, ldb, lane);
    v16bf b1 = load_frag_b(Bbase1 + k0, ldb, lane);
    acc00 = wmma_bf16(a0, b0, acc00);
    acc01 = wmma_bf16(a0, b1, acc01);
    acc10 = wmma_bf16(a1, b0, acc10);
    acc11 = wmma_bf16(a1, b1, acc11);
  }
  store_tile(C, ldc, m0,      n0,      acc00, bias, alpha, lane);
  store_tile(C, ldc, m0,      n0 + 16, acc01, bias, alpha, lane);
  store_tile(C, ldc, m0 + 16, n0,      acc10, bias, alpha, lane);
  store_tile(C, ldc, m0 + 16, n0 + 16, acc11, bias, alpha, lane);
}

// ---------------- helpers: f32 -> bf16 convert, transpose-convert ----------------
__global__ __launch_bounds__(256) void convert_f32_bf16(const float* __restrict__ in,
                                                        bf16_t* __restrict__ out, int n) {
  int i = blockIdx.x * blockDim.x + threadIdx.x;
  if (i < n) out[i] = (bf16_t)in[i];
}

// in: f32 [R x C] row-major -> out: bf16 [C x R] row-major
__global__ __launch_bounds__(256) void transpose_convert(const float* __restrict__ in,
                                                         bf16_t* __restrict__ out, int R, int C) {
  __shared__ float tile[32][33];
  int ctiles = C >> 5;
  int bx = blockIdx.x % ctiles, by = blockIdx.x / ctiles;
  int c0 = bx << 5, r0 = by << 5;
  int tx = threadIdx.x & 31, ty = threadIdx.x >> 5;   // 32 x 8
#pragma unroll
  for (int i = 0; i < 32; i += 8)
    tile[ty + i][tx] = in[(size_t)(r0 + ty + i) * C + c0 + tx];
  __syncthreads();
#pragma unroll
  for (int i = 0; i < 32; i += 8)
    out[(size_t)(c0 + ty + i) * R + r0 + tx] = (bf16_t)tile[tx][ty + i];
}

// ---------------- GRU recurrence: one block per direction, h in LDS ----------------
__global__ __launch_bounds__(512) void gru_scan_kernel(
    const float* __restrict__ gx_base,
    const float* __restrict__ Whh0, const float* __restrict__ Whh1,
    const float* __restrict__ Whh2, const float* __restrict__ Whh3,
    const float* __restrict__ bhh0, const float* __restrict__ bhh1,
    const float* __restrict__ bhh2, const float* __restrict__ bhh3,
    float* __restrict__ srcf, float* __restrict__ tgtf, int L)
{
  __shared__ __align__(16) float hs[HDIM];
  int d = blockIdx.x;
  int tid = threadIdx.x;
  const float* gx  = gx_base + (size_t)d * L * G3;
  const float* Whh = (d == 0) ? Whh0 : (d == 1) ? Whh1 : (d == 2) ? Whh2 : Whh3;
  const float* bhh = (d == 0) ? bhh0 : (d == 1) ? bhh1 : (d == 2) ? bhh2 : bhh3;
  float* out = ((d < 2) ? srcf : tgtf) + ((d & 1) ? HDIM : 0);  // fwd -> cols 0..H-1, bwd -> H..2H-1
  int rev = d & 1;

  hs[tid] = 0.0f;
  __syncthreads();

  const float4* Wr = (const float4*)(Whh + (size_t)(0 * HDIM + tid) * HDIM);
  const float4* Wz = (const float4*)(Whh + (size_t)(1 * HDIM + tid) * HDIM);
  const float4* Wn = (const float4*)(Whh + (size_t)(2 * HDIM + tid) * HDIM);
  float br = bhh[tid], bz = bhh[HDIM + tid], bn = bhh[2 * HDIM + tid];

  for (int t = 0; t < L; ++t) {
    int idx = rev ? (L - 1 - t) : t;
    float ar = br, az = bz, an = bn;
    const float4* h4 = (const float4*)hs;
#pragma unroll 4
    for (int k = 0; k < HDIM / 4; ++k) {
      float4 h = h4[k], wr = Wr[k], wz = Wz[k], wn = Wn[k];
      ar += h.x * wr.x + h.y * wr.y + h.z * wr.z + h.w * wr.w;
      az += h.x * wz.x + h.y * wz.y + h.z * wz.z + h.w * wz.w;
      an += h.x * wn.x + h.y * wn.y + h.z * wn.z + h.w * wn.w;
    }
    const float* g = gx + (size_t)idx * G3;
    float r = 1.0f / (1.0f + __expf(-(g[tid] + ar)));
    float z = 1.0f / (1.0f + __expf(-(g[HDIM + tid] + az)));
    float n = tanhf(g[2 * HDIM + tid] + r * an);
    float hn = (1.0f - z) * n + z * hs[tid];
    out[(size_t)idx * D2 + tid] = hn;
    __syncthreads();
    hs[tid] = hn;
    __syncthreads();
  }
}

// ---------------- row softmax (optional diagonal mask) -> bf16 probabilities ----------------
__global__ __launch_bounds__(256) void softmax_row_kernel(float* __restrict__ S,
                                                          bf16_t* __restrict__ P,
                                                          int C, int maskdiag) {
  __shared__ float red[256];
  int row = blockIdx.x, tid = threadIdx.x;
  float* srow = S + (size_t)row * C;
  float m = -3.4e38f;
  for (int c = tid; c < C; c += 256) {
    float v = srow[c];
    if (maskdiag && c == row) v = -3.4e38f;
    m = fmaxf(m, v);
  }
  red[tid] = m; __syncthreads();
  for (int s = 128; s > 0; s >>= 1) { if (tid < s) red[tid] = fmaxf(red[tid], red[tid + s]); __syncthreads(); }
  m = red[0]; __syncthreads();
  float sum = 0.0f;
  for (int c = tid; c < C; c += 256) {
    float v = srow[c];
    if (maskdiag && c == row) v = -3.4e38f;
    float e = __expf(v - m);
    srow[c] = e;
    sum += e;
  }
  red[tid] = sum; __syncthreads();
  for (int s = 128; s > 0; s >>= 1) { if (tid < s) red[tid] += red[tid + s]; __syncthreads(); }
  float inv = 1.0f / red[0];
  for (int c = tid; c < C; c += 256) P[(size_t)row * C + c] = (bf16_t)(srow[c] * inv);
}

// ---------------- emit: feats[T,6H] . W_emit[2,6H]^T + b ----------------
__global__ __launch_bounds__(256) void emit_kernel(const float* __restrict__ tgtf,
                                                   const float* __restrict__ ctx,
                                                   const float* __restrict__ sctx,
                                                   const float* __restrict__ Wm,
                                                   const float* __restrict__ bm,
                                                   float* __restrict__ emit, int T) {
  int t = blockIdx.x * blockDim.x + threadIdx.x;
  if (t >= T) return;
  const float* f0 = tgtf + (size_t)t * D2;
  const float* f1 = ctx  + (size_t)t * D2;
  const float* f2 = sctx + (size_t)t * D2;
#pragma unroll
  for (int c = 0; c < 2; ++c) {
    const float* w = Wm + (size_t)c * (3 * D2);
    float s = bm[c];
    for (int k = 0; k < D2; ++k) s += f0[k] * w[k];
    for (int k = 0; k < D2; ++k) s += f1[k] * w[D2 + k];
    for (int k = 0; k < D2; ++k) s += f2[k] * w[2 * D2 + k];
    emit[t * 2 + c] = s;
  }
}

// ---------------- CRF: gold path score - logZ (NT = 2) ----------------
__global__ void crf_kernel(const float* __restrict__ emit, const int* __restrict__ labels,
                           const float* __restrict__ t_start, const float* __restrict__ t_trans,
                           const float* __restrict__ t_end, float* __restrict__ out, int T) {
  if (threadIdx.x != 0 || blockIdx.x != 0) return;
  float gold = t_start[labels[0]];
  for (int t = 0; t < T; ++t) gold += emit[t * 2 + labels[t]];
  for (int t = 1; t < T; ++t) gold += t_trans[labels[t] * 2 + labels[t - 1]];
  gold += t_end[labels[T - 1]];

  float a0 = t_start[0] + emit[0];
  float a1 = t_start[1] + emit[1];
  for (int t = 1; t < T; ++t) {
    float x00 = t_trans[0] + a0, x01 = t_trans[1] + a1;
    float m0 = fmaxf(x00, x01);
    float n0 = m0 + logf(expf(x00 - m0) + expf(x01 - m0)) + emit[t * 2 + 0];
    float x10 = t_trans[2] + a0, x11 = t_trans[3] + a1;
    float m1 = fmaxf(x10, x11);
    float n1 = m1 + logf(expf(x10 - m1) + expf(x11 - m1)) + emit[t * 2 + 1];
    a0 = n0; a1 = n1;
  }
  float e0 = t_end[0] + a0, e1 = t_end[1] + a1;
  float mm = fmaxf(e0, e1);
  float logZ = mm + logf(expf(e0 - mm) + expf(e1 - mm));
  out[0] = gold - logZ;
}

// ---------------- host side ----------------
static void launch_gemm(const bf16_t* A, const bf16_t* B, float* C, const float* bias,
                        int M, int N, int K, int lda, int ldb, int ldc, float alpha,
                        hipStream_t stream) {
  int waves = (M >> 5) * (N >> 5);
  int blocks = (waves + 7) >> 3;            // 8 waves (256 threads) per block
  wmma_gemm_bf16_kernel<<<blocks, 256, 0, stream>>>(A, B, C, bias, M, N, K, lda, ldb, ldc, alpha);
}

extern "C" void kernel_launch(void* const* d_in, const int* in_sizes, int n_in,
                              void* d_out, int out_size, void* d_ws, size_t ws_size,
                              hipStream_t stream) {
  (void)in_sizes; (void)n_in; (void)out_size; (void)ws_size;
  const float* source = (const float*)d_in[0];
  const float* target = (const float*)d_in[1];
  const int*   labels = (const int*)d_in[2];
  const float* Wih[4] = { (const float*)d_in[3],  (const float*)d_in[7],
                          (const float*)d_in[11], (const float*)d_in[15] };
  const float* Whh[4] = { (const float*)d_in[4],  (const float*)d_in[8],
                          (const float*)d_in[12], (const float*)d_in[16] };
  const float* bih[4] = { (const float*)d_in[5],  (const float*)d_in[9],
                          (const float*)d_in[13], (const float*)d_in[17] };
  const float* bhh[4] = { (const float*)d_in[6],  (const float*)d_in[10],
                          (const float*)d_in[14], (const float*)d_in[18] };
  const float* W_emit  = (const float*)d_in[19];
  const float* b_emit  = (const float*)d_in[20];
  const float* t_start = (const float*)d_in[21];
  const float* t_trans = (const float*)d_in[22];
  const float* t_end   = (const float*)d_in[23];

  // workspace arena (~80 MB peak, gx region recycled after the recurrent scan)
  const size_t MB = 1024 * 1024;
  char* W = (char*)d_ws;
  float*  gx_base  = (float*)(W);                 // 48MB: 4 x [2048x1536] f32 (live: gemm->scan)
  float*  scores   = (float*)(W);                 // 16MB  (aliases gx, live post-scan)
  bf16_t* probs    = (bf16_t*)(W + 16 * MB);      //  8MB  (aliases gx)
  float*  ctx      = (float*)(W + 24 * MB);       //  8MB  (aliases gx)
  float*  sctx     = (float*)(W + 32 * MB);       //  8MB  (aliases gx)
  float*  srcf     = (float*)(W + 48 * MB);       //  8MB
  float*  tgtf     = (float*)(W + 56 * MB);       //  8MB
  bf16_t* srcf_bf  = (bf16_t*)(W + 64 * MB);      //  4MB (post-scan)
  bf16_t* tgtf_bf  = (bf16_t*)(W + 68 * MB);      //  4MB (post-scan)
  bf16_t* srcfT_bf = (bf16_t*)(W + 72 * MB);      //  4MB (post-scan)
  bf16_t* tgtfT_bf = (bf16_t*)(W + 76 * MB);      //  4MB (post-scan)
  bf16_t* xsrc_bf  = (bf16_t*)(W + 64 * MB);      //  2MB (pre-scan only; aliases srcf_bf)
  bf16_t* xtgt_bf  = (bf16_t*)(W + 66 * MB);      //  2MB (pre-scan only)
  bf16_t* wih_bf   = (bf16_t*)(W + 68 * MB);      //  6MB (pre-scan only; aliases tgtf_bf/srcfT_bf)
  float*  emitb    = (float*)(W + 80 * MB);       // 16KB

  const int CVT = 256;

  // 1) bf16 copies of inputs + input-gate weights
  convert_f32_bf16<<<(SLEN * EDIM + CVT - 1) / CVT, CVT, 0, stream>>>(source, xsrc_bf, SLEN * EDIM);
  convert_f32_bf16<<<(TLEN * EDIM + CVT - 1) / CVT, CVT, 0, stream>>>(target, xtgt_bf, TLEN * EDIM);
  for (int d = 0; d < 4; ++d)
    convert_f32_bf16<<<(G3 * EDIM + CVT - 1) / CVT, CVT, 0, stream>>>(
        Wih[d], wih_bf + (size_t)d * G3 * EDIM, G3 * EDIM);

  // 2) gx[d] = x @ Wih[d]^T + bih[d]   (WMMA, M=2048 N=1536 K=512)
  for (int d = 0; d < 4; ++d) {
    const bf16_t* x = (d < 2) ? xsrc_bf : xtgt_bf;
    launch_gemm(x, wih_bf + (size_t)d * G3 * EDIM, gx_base + (size_t)d * SLEN * G3, bih[d],
                SLEN, G3, EDIM, EDIM, EDIM, G3, 1.0f, stream);
  }

  // 3) sequential GRU recurrence, 4 directions in parallel (one WGP each)
  gru_scan_kernel<<<4, 512, 0, stream>>>(gx_base,
                                         Whh[0], Whh[1], Whh[2], Whh[3],
                                         bhh[0], bhh[1], bhh[2], bhh[3],
                                         srcf, tgtf, SLEN);

  // 4) bf16 + transposed-bf16 copies of encoder outputs for attention WMMA
  convert_f32_bf16<<<(SLEN * D2 + CVT - 1) / CVT, CVT, 0, stream>>>(srcf, srcf_bf, SLEN * D2);
  convert_f32_bf16<<<(TLEN * D2 + CVT - 1) / CVT, CVT, 0, stream>>>(tgtf, tgtf_bf, TLEN * D2);
  transpose_convert<<<(SLEN / 32) * (D2 / 32), 256, 0, stream>>>(srcf, srcfT_bf, SLEN, D2);
  transpose_convert<<<(TLEN / 32) * (D2 / 32), 256, 0, stream>>>(tgtf, tgtfT_bf, TLEN, D2);

  const float inv_temp = 32.0f;  // 1/temp = sqrt(2H)

  // 5) cross attention: scores = (tgt_f . src_f^T)/temp ; softmax ; ctx = P . src_f
  launch_gemm(tgtf_bf, srcf_bf, scores, nullptr, TLEN, SLEN, D2, D2, D2, SLEN, inv_temp, stream);
  softmax_row_kernel<<<TLEN, 256, 0, stream>>>(scores, probs, SLEN, 0);
  launch_gemm(probs, srcfT_bf, ctx, nullptr, TLEN, D2, SLEN, SLEN, SLEN, D2, 1.0f, stream);

  // 6) self attention with diagonal mask
  launch_gemm(tgtf_bf, tgtf_bf, scores, nullptr, TLEN, TLEN, D2, D2, D2, TLEN, inv_temp, stream);
  softmax_row_kernel<<<TLEN, 256, 0, stream>>>(scores, probs, TLEN, 1);
  launch_gemm(probs, tgtfT_bf, sctx, nullptr, TLEN, D2, TLEN, TLEN, TLEN, D2, 1.0f, stream);

  // 7) emissions (NT=2 -> plain VALU)
  emit_kernel<<<(TLEN + 255) / 256, 256, 0, stream>>>(tgtf, ctx, sctx, W_emit, b_emit, emitb, TLEN);

  // 8) CRF gold score - log partition
  crf_kernel<<<1, 32, 0, stream>>>(emitb, labels, t_start, t_trans, t_end, (float*)d_out, TLEN);
}
